// MaxOccurrencePool_63127429316885
// MI455X (gfx1250) — matrix operand order repaired and checked by reading, not compiled, not run on hardware
//
#include <hip/hip_runtime.h>

// MaxOccurrencePool on MI455X (gfx1250, wave32).
// Memory-bound: 64MB in + 128MB out -> ~8.2us floor @ 23.3TB/s.
// Window-histogram computed as band-matrix x one-hot via v_wmma_f32_16x16x32_f16.
// Input staged to LDS via GLOBAL_LOAD_ASYNC_TO_LDS_B128 (ASYNCcnt-tracked).

typedef __attribute__((ext_vector_type(16))) _Float16 v16h;
typedef __attribute__((ext_vector_type(8)))  float    v8f;
typedef __attribute__((ext_vector_type(4)))  float    vf4;
typedef __attribute__((ext_vector_type(4)))  int      v4i;

#define B_DIM 8
#define L_DIM 131072
#define C_DIM 16
#define TILE_L 128              // per workgroup (8 waves x 16 rows)
#define SLAB   (TILE_L + 8)     // l0-3 .. l0+132 -> 136 positions
#define CNT_S  17               // padded stride (dwords) for count spill

#if defined(__AMDGCN__) && __has_builtin(__builtin_amdgcn_global_load_async_to_lds_b128)
#define HAVE_ASYNC 1
typedef __attribute__((address_space(1))) v4i gv4i;   // global-AS int4
typedef __attribute__((address_space(3))) v4i lv4i;   // LDS-AS int4
#else
#define HAVE_ASYNC 0
#endif

__global__ __launch_bounds__(256)
void mop_wmma_kernel(const float* __restrict__ x, float* __restrict__ out)
{
    __shared__ float lds_x[SLAB * C_DIM];        // 2176 dwords
    __shared__ float lds_cnt[8 * 16 * CNT_S];    // per-wave 16x17 spill of D

    const int tid  = threadIdx.x;
    const int b    = blockIdx.y;
    const int l0   = blockIdx.x * TILE_L;
    const int w    = tid >> 5;                   // wave id (wave32)
    const int lane = tid & 31;

    // ---- stage slab [l0-3, l0+132] x 16 channels, coalesced, zero-padded ----
#if HAVE_ASYNC
    // 16B per lane, direct global->LDS (no VGPR round-trip), ASYNCcnt-tracked.
    for (int i = tid; i < (SLAB * C_DIM) / 4; i += 256) {
        const int l = l0 - 3 + (i >> 2);         // each i covers one 16B quad
        if (l >= 0 && l < L_DIM) {
            const float* g = x + ((size_t)b * L_DIM + l) * C_DIM + (size_t)(i & 3) * 4;
            __builtin_amdgcn_global_load_async_to_lds_b128(
                (gv4i*)(v4i*)g,
                (lv4i*)(v4i*)&lds_x[4 * i],
                0, 0);
        } else {
            vf4 z = {0.0f, 0.0f, 0.0f, 0.0f};
            *(vf4*)&lds_x[4 * i] = z;
        }
    }
#if __has_builtin(__builtin_amdgcn_s_wait_asynccnt)
    __builtin_amdgcn_s_wait_asynccnt(0);
#else
    asm volatile("s_wait_asynccnt 0" ::: "memory");
#endif
#else
    for (int i = tid; i < SLAB * C_DIM; i += 256) {
        const int ls = i >> 4;
        const int cc = i & 15;
        const int l  = l0 - 3 + ls;
        float v = 0.0f;
        if (l >= 0 && l < L_DIM)
            v = __builtin_nontemporal_load(&x[((size_t)b * L_DIM + l) * C_DIM + cc]);
        lds_x[i] = v;
    }
#endif
    __syncthreads();

    // ---- constant band matrix A[m,k] = (m <= k <= m+8), 16x32 f16 ----
    // 16-bit A layout: lane<16 -> K = {0..7, 16..23}; lane>=16 -> K = {8..15, 24..31}
    const int  m  = lane & 15;
    const bool lo = lane < 16;
    v16h amat;
#pragma unroll
    for (int h = 0; h < 16; ++h) {
        const int K = (h < 8) ? (lo ? h : h + 8) : (lo ? h + 8 : h + 16);
        amat[h] = (_Float16)(((K >= m) && (K <= m + 8)) ? 1.0f : 0.0f);
    }

    // B layout (32x16): lane = column n (lane&15); lane<16 holds K={0..7,16..23}, lane>=16 K={8..15,24..31}
    const int   n     = lane & 15;
    const bool  coln  = (n >= 1) && (n <= 4);    // only value bins 1..4 live
    const float nf    = (float)n;
    const int   kbase = lo ? 0 : 16;
    const int   row   = lane & 15;
    float* cw = &lds_cnt[w * 16 * CNT_S];

    float res[16];

#pragma unroll
    for (int c = 0; c < C_DIM; ++c) {
        // ---- build one-hot B for this channel from LDS (broadcast reads) ----
        v16h bmat;
#pragma unroll
        for (int h = 0; h < 16; ++h) {
            const int k = (kbase >> 1) + h + (h < 8 ? 0 : 8);             // B row index 0..31
            float xv = 0.0f;
            if (k < 24) xv = lds_x[(16 * w + k) * C_DIM + c];
            bmat[h] = (_Float16)(((k < 24) && coln && (xv == nf)) ? 1.0f : 0.0f);
        }

        // ---- counts[m, v] = sum over window of [x==v]  (exact small ints) ----
        v8f cz = {};
        v8f acc = __builtin_amdgcn_wmma_f32_16x16x32_f16(
            /*neg_a=*/false, amat, /*neg_b=*/false, bmat,
            /*c_mod=*/(short)0, cz, /*reuse_a=*/false, /*reuse_b=*/false);

        // ---- spill D to per-wave LDS: lane holds col n, VGPR i = row m0+i ----
        const int m0 = lo ? 0 : 8;
#pragma unroll
        for (int i = 0; i < 8; ++i)
            cw[(m0 + i) * CNT_S + n] = acc[i];

        __builtin_amdgcn_wave_barrier();
        asm volatile("s_wait_dscnt 0" ::: "memory");  // same-wave LDS RAW

        // ---- tie-averaged argmax over bins (bin0 analytic: 1 iff sum==0) ----
        const float c1 = cw[row * CNT_S + 1];
        const float c2 = cw[row * CNT_S + 2];
        const float c3 = cw[row * CNT_S + 3];
        const float c4 = cw[row * CNT_S + 4];
        const float s  = c1 + c2 + c3 + c4;
        const float mx = fmaxf(fmaxf(c1, c2), fmaxf(c3, c4));
        const float t  = (c1 == mx ? 1.f : 0.f) + (c2 == mx ? 1.f : 0.f)
                       + (c3 == mx ? 1.f : 0.f) + (c4 == mx ? 1.f : 0.f);
        const float vs = (c1 == mx ? 1.f : 0.f) + (c2 == mx ? 2.f : 0.f)
                       + (c3 == mx ? 3.f : 0.f) + (c4 == mx ? 4.f : 0.f);
        // soft-argmax limit: uniform weight over tied maxima -> vs * (1/t).
        // t in {1,2,3,4}: v_rcp_f32 is exact for 1,2,4 and ~1ulp for 3.
        const float ov = (s == 0.0f) ? 0.0f : vs * __builtin_amdgcn_rcpf(t);
        const float dv = (s == 0.0f) ? 1.0f : mx;       // max count
        res[c] = lo ? ov : dv;

        __builtin_amdgcn_wave_barrier();
    }

    // ---- coalesced epilogue: each lane owns one full 64B output line ----
    // lanes 0-15 -> out[b, lw0+row, 0..15]; lanes 16-31 -> density line
    const size_t outN = (size_t)B_DIM * L_DIM * C_DIM;
    const size_t base = ((size_t)b * L_DIM + (size_t)(l0 + 16 * w)) * C_DIM;
    float* p = out + (lo ? 0 : outN) + base + (size_t)row * C_DIM;
#pragma unroll
    for (int q = 0; q < 4; ++q) {
        vf4 v = { res[4 * q + 0], res[4 * q + 1], res[4 * q + 2], res[4 * q + 3] };
        __builtin_nontemporal_store(v, (vf4*)(p + 4 * q));
    }
}

extern "C" void kernel_launch(void* const* d_in, const int* in_sizes, int n_in,
                              void* d_out, int out_size, void* d_ws, size_t ws_size,
                              hipStream_t stream) {
    (void)in_sizes; (void)n_in; (void)d_ws; (void)ws_size; (void)out_size;
    const float* x = (const float*)d_in[0];
    float* out = (float*)d_out;
    dim3 grid(L_DIM / TILE_L, B_DIM);   // 1024 x 8 workgroups
    dim3 block(256);                    // 8 wave32 waves
    hipLaunchKernelGGL(mop_wmma_kernel, grid, block, 0, stream, x, out);
}
